// MoELayer_12678743458186
// MI455X (gfx1250) — compile-verified, matrix operand order checked by reading, not run on hardware
//
#include <hip/hip_runtime.h>

// MoE (top-2 of 8 experts) for gfx1250 / MI455X.
// B=2, S=2048 -> T=4096 tokens, H=1024, I=2816, E=8, K=2.
// fp32->bf16 cast pass (one HBM sweep, ~300MB @ 23.3TB/s), wave32 router with
// top-2 gather-by-expert, then per-(expert, 16-token-tile) fused SwiGLU+down
// GEMM on v_wmma_f32_16x16x32_bf16. The 16x2816 bf16 intermediate lives in
// the 320KB-per-WGP LDS; X tile staged with CDNA5 async global->LDS copies
// (ASYNCcnt). bf16 weights (138MB) fit the 192MB L2, so the compute phase
// streams B operands from L2. Output committed with f32 global atomics
// (exactly two commutative adds per element -> bitwise deterministic).

#define T_TOK 4096
#define H_DIM 1024
#define I_DIM 2816
#define E_NUM 8

typedef __attribute__((ext_vector_type(16))) __bf16 v16bf;
typedef __attribute__((ext_vector_type(8)))  __bf16 v8bf;
typedef __attribute__((ext_vector_type(8)))  float  v8f;

__device__ __forceinline__ unsigned short f2bf(float f) {
  unsigned u = __float_as_uint(f);
  u += 0x7FFFu + ((u >> 16) & 1u);   // round-to-nearest-even
  return (unsigned short)(u >> 16);
}

__device__ __forceinline__ v16bf cat8(v8bf lo, v8bf hi) {
  return __builtin_shufflevector(lo, hi, 0,1,2,3,4,5,6,7,8,9,10,11,12,13,14,15);
}

// ---------------------------------------------------------------------------
// Zero output, write balance loss (== E exactly: softmax probs > 0 always, so
// the indicator mean is 1 and sum_e avg_probs_e == 1), zero expert counters.
// ---------------------------------------------------------------------------
__global__ void moe_init_kernel(float* __restrict__ out, int* __restrict__ counts) {
  long i = (long)blockIdx.x * blockDim.x + threadIdx.x;
  long stride = (long)gridDim.x * blockDim.x;
  for (long j = i; j < (long)T_TOK * H_DIM; j += stride) out[j] = 0.0f;
  if (i == 0) out[(long)T_TOK * H_DIM] = (float)E_NUM;  // balance loss
  if (i < E_NUM) counts[i] = 0;
}

// ---------------------------------------------------------------------------
// fp32 -> bf16 streaming cast (one HBM pass over x + weights).
// ---------------------------------------------------------------------------
__global__ void cast_f32_bf16_kernel(const float* __restrict__ in,
                                     unsigned short* __restrict__ out, long n) {
  long i = (long)blockIdx.x * blockDim.x + threadIdx.x;
  long stride = (long)gridDim.x * blockDim.x;
  for (; i < n; i += stride) out[i] = f2bf(in[i]);
}

// ---------------------------------------------------------------------------
// Router: one wave32 per token. 8 H=1024 dot products, softmax top-2
// (denominator cancels under top-k renorm), scatter into per-expert lists.
// ---------------------------------------------------------------------------
__global__ void router_kernel(const float* __restrict__ x,
                              const float* __restrict__ gw,
                              int* __restrict__ counts,
                              int* __restrict__ tok_list,
                              float* __restrict__ wt_list) {
  int wave = threadIdx.x >> 5;
  int lane = threadIdx.x & 31;
  int t = blockIdx.x * (blockDim.x >> 5) + wave;
  if (t >= T_TOK) return;

  float acc[E_NUM];
#pragma unroll
  for (int e = 0; e < E_NUM; ++e) acc[e] = 0.0f;

  for (int h = lane; h < H_DIM; h += 32) {
    float xv = x[(long)t * H_DIM + h];
#pragma unroll
    for (int e = 0; e < E_NUM; ++e) acc[e] += xv * gw[e * H_DIM + h];
  }
#pragma unroll
  for (int e = 0; e < E_NUM; ++e) {
#pragma unroll
    for (int m = 16; m >= 1; m >>= 1) acc[e] += __shfl_xor(acc[e], m, 32);
  }

  if (lane == 0) {
    float mx = acc[0];
#pragma unroll
    for (int e = 1; e < E_NUM; ++e) mx = fmaxf(mx, acc[e]);
    float p[E_NUM];
#pragma unroll
    for (int e = 0; e < E_NUM; ++e) p[e] = __expf(acc[e] - mx);

    int e0 = 0;
#pragma unroll
    for (int e = 1; e < E_NUM; ++e) if (p[e] > p[e0]) e0 = e;
    int e1 = (e0 == 0) ? 1 : 0;
#pragma unroll
    for (int e = 0; e < E_NUM; ++e) if (e != e0 && p[e] > p[e1]) e1 = e;

    float s = p[e0] + p[e1];
    float w0 = p[e0] / s, w1 = p[e1] / s;

    int s0 = atomicAdd(&counts[e0], 1);
    tok_list[e0 * T_TOK + s0] = t;
    wt_list[e0 * T_TOK + s0] = w0;
    int s1 = atomicAdd(&counts[e1], 1);
    tok_list[e1 * T_TOK + s1] = t;
    wt_list[e1 * T_TOK + s1] = w1;
  }
}

// ---------------------------------------------------------------------------
// Fused expert FFN: per (expert, 16-token tile) block, 8 waves.
// Phase 1: h[16, I] = silu(X Gp^T) * (X Up^T) in LDS (bf16).
// Phase 2: out += diag(w) * h Dp^T via WMMA + f32 atomics.
// ---------------------------------------------------------------------------
__global__ __launch_bounds__(256)
void moe_ffn_kernel(const unsigned short* __restrict__ x_bf,
                    const unsigned short* __restrict__ gp_bf,
                    const unsigned short* __restrict__ up_bf,
                    const unsigned short* __restrict__ dp_bf,
                    const int* __restrict__ counts,
                    const int* __restrict__ tok_list,
                    const float* __restrict__ wt_list,
                    float* __restrict__ out) {
  __shared__ __align__(16) __bf16 h_s[16 * I_DIM];   // 90112 B
  __shared__ __align__(16) __bf16 x_s[16 * H_DIM];   // 32768 B
  __shared__ int   tok_s[16];
  __shared__ float wt_s[16];

  const int tilesPerExpert = T_TOK / 16;             // 256
  const int e    = blockIdx.x / tilesPerExpert;
  const int tile = blockIdx.x % tilesPerExpert;
  const int cnt  = counts[e];
  const int base = tile * 16;
  if (base >= cnt) return;

  const int tid = threadIdx.x;
  if (tid < 16) {
    int r = base + tid;
    if (r < cnt) {
      tok_s[tid] = tok_list[e * T_TOK + r];
      wt_s[tid]  = wt_list[e * T_TOK + r];
    } else {                       // pad: valid token, zero weight
      tok_s[tid] = tok_list[e * T_TOK];
      wt_s[tid]  = 0.0f;
    }
  }
  __syncthreads();

  // Stage gathered X rows (16 x 1024 bf16 = 32KB) into LDS via CDNA5 async
  // global->LDS b128 copies (ASYNCcnt path, no VGPR round-trip).
  {
    int r = tid >> 4;              // row 0..15
    int c = tid & 15;              // chunk 0..15, 8 x b128 each
    const char* src = (const char*)(x_bf + (size_t)tok_s[r] * H_DIM) + c * 128;
    unsigned dst = (unsigned)(unsigned long long)(x_s + r * H_DIM) + (unsigned)(c * 128);
#pragma unroll
    for (int q = 0; q < 8; ++q) {
      asm volatile("global_load_async_to_lds_b128 %0, %1, off"
                   :: "v"(dst + (unsigned)(q * 16)), "v"(src + q * 16)
                   : "memory");
    }
#if __has_builtin(__builtin_amdgcn_s_wait_asynccnt)
    __builtin_amdgcn_s_wait_asynccnt(0);
#else
    asm volatile("s_wait_asynccnt 0" ::: "memory");
#endif
  }
  __syncthreads();

  const int wave  = tid >> 5;
  const int lane  = tid & 31;
  const int row   = lane & 15;     // A-matrix M / B-matrix N owned by lane
  const int khalf = lane >> 4;     // K-half per ISA 16-bit operand striping

  // -------- Phase 1: h = silu(X Gp^T) * (X Up^T), wave covers 22 N-tiles.
  // Gate/up share the A fragment -> 2 WMMA per LDS A fetch.
  for (int nt = 0; nt < 22; ++nt) {
    const int ncol = (wave * 22 + nt) * 16 + row;          // I column
    const unsigned short* gB = gp_bf + ((size_t)e * I_DIM + ncol) * H_DIM;
    const unsigned short* uB = up_bf + ((size_t)e * I_DIM + ncol) * H_DIM;
    v8f accg = {};
    v8f accu = {};
    for (int k0 = 0; k0 < H_DIM; k0 += 32) {
      const int ka = k0 + khalf * 8;
      v16bf a = cat8(*(const v8bf*)(x_s + row * H_DIM + ka),
                     *(const v8bf*)(x_s + row * H_DIM + ka + 16));
      v16bf bg = cat8(*(const v8bf*)(gB + ka), *(const v8bf*)(gB + ka + 16));
      v16bf bu = cat8(*(const v8bf*)(uB + ka), *(const v8bf*)(uB + ka + 16));
      accg = __builtin_amdgcn_wmma_f32_16x16x32_bf16(false, a, false, bg,
                                                     (short)0, accg, false, false);
      accu = __builtin_amdgcn_wmma_f32_16x16x32_bf16(false, a, false, bu,
                                                     (short)0, accu, false, false);
    }
    unsigned short* hw = (unsigned short*)h_s;
#pragma unroll
    for (int r2 = 0; r2 < 8; ++r2) {
      const int m = r2 + khalf * 8;                        // token row in tile
      float g = accg[r2];
      float u = accu[r2];
      float h = (g / (1.0f + __expf(-g))) * u;             // silu(g) * u
      hw[m * I_DIM + ncol] = f2bf(h);
    }
  }
  __syncthreads();

  // -------- Phase 2: out_tile = diag(w) * h Dp^T. Wave covers 8 N-tiles of
  // H, processed in pairs so each LDS A fragment feeds 2 WMMA.
  for (int nt = 0; nt < 4; ++nt) {
    const int ncol0 = (wave * 8 + 2 * nt) * 16 + row;      // H column, tile 0
    const int ncol1 = ncol0 + 16;                          // H column, tile 1
    const unsigned short* dB0 = dp_bf + ((size_t)e * H_DIM + ncol0) * I_DIM;
    const unsigned short* dB1 = dp_bf + ((size_t)e * H_DIM + ncol1) * I_DIM;
    v8f acc0 = {};
    v8f acc1 = {};
    for (int k0 = 0; k0 < I_DIM; k0 += 32) {
      const int ka = k0 + khalf * 8;
      __builtin_prefetch(dB0 + ka + 512, 0, 1);            // global_prefetch_b8
      __builtin_prefetch(dB1 + ka + 512, 0, 1);
      v16bf a  = cat8(*(const v8bf*)(h_s + row * I_DIM + ka),
                      *(const v8bf*)(h_s + row * I_DIM + ka + 16));
      v16bf b0 = cat8(*(const v8bf*)(dB0 + ka), *(const v8bf*)(dB0 + ka + 16));
      v16bf b1 = cat8(*(const v8bf*)(dB1 + ka), *(const v8bf*)(dB1 + ka + 16));
      acc0 = __builtin_amdgcn_wmma_f32_16x16x32_bf16(false, a, false, b0,
                                                     (short)0, acc0, false, false);
      acc1 = __builtin_amdgcn_wmma_f32_16x16x32_bf16(false, a, false, b1,
                                                     (short)0, acc1, false, false);
    }
#pragma unroll
    for (int r2 = 0; r2 < 8; ++r2) {
      const int m = r2 + khalf * 8;
      float w = wt_s[m];
      size_t rowoff = (size_t)tok_s[m] * H_DIM;
      atomicAdd(out + rowoff + ncol0, acc0[r2] * w);
      atomicAdd(out + rowoff + ncol1, acc1[r2] * w);
    }
  }
}

// ---------------------------------------------------------------------------
extern "C" void kernel_launch(void* const* d_in, const int* in_sizes, int n_in,
                              void* d_out, int out_size, void* d_ws, size_t ws_size,
                              hipStream_t stream) {
  const float* x         = (const float*)d_in[0];  // [T, H]
  const float* gate_w    = (const float*)d_in[1];  // [E, H]
  const float* gate_proj = (const float*)d_in[2];  // [E, I, H]
  const float* up_proj   = (const float*)d_in[3];  // [E, I, H]
  const float* down_proj = (const float*)d_in[4];  // [E, H, I]
  float* out = (float*)d_out;                      // [T, H] + 1 (balance loss)

  const size_t TH  = (size_t)T_TOK * H_DIM;
  const size_t EIH = (size_t)E_NUM * I_DIM * H_DIM;

  char* ws = (char*)d_ws;
  unsigned short* x_bf  = (unsigned short*)ws; ws += TH * 2;
  unsigned short* gp_bf = (unsigned short*)ws; ws += EIH * 2;
  unsigned short* up_bf = (unsigned short*)ws; ws += EIH * 2;
  unsigned short* dp_bf = (unsigned short*)ws; ws += EIH * 2;
  int*   counts   = (int*)ws;   ws += 256;         // padded
  int*   tok_list = (int*)ws;   ws += (size_t)E_NUM * T_TOK * 4;
  float* wt_list  = (float*)ws; ws += (size_t)E_NUM * T_TOK * 4;

  moe_init_kernel<<<1024, 256, 0, stream>>>(out, counts);

  cast_f32_bf16_kernel<<<2048, 256, 0, stream>>>(x, x_bf, (long)TH);
  cast_f32_bf16_kernel<<<4096, 256, 0, stream>>>(gate_proj, gp_bf, (long)EIH);
  cast_f32_bf16_kernel<<<4096, 256, 0, stream>>>(up_proj,   up_bf, (long)EIH);
  cast_f32_bf16_kernel<<<4096, 256, 0, stream>>>(down_proj, dp_bf, (long)EIH);

  router_kernel<<<T_TOK / 8, 256, 0, stream>>>(x, gate_w, counts, tok_list, wt_list);

  moe_ffn_kernel<<<E_NUM * (T_TOK / 16), 256, 0, stream>>>(
      x_bf, gp_bf, up_bf, dp_bf, counts, tok_list, wt_list, out);
}
